// HeteroGATLayerReal_84404697301142
// MI455X (gfx1250) — matrix operand order repaired and checked by reading, not compiled, not run on hardware
//
#include <hip/hip_runtime.h>
#include <hip/hip_bf16.h>
#include <math.h>

#define H   4
#define DD  32
#define HD  128     // H*D
#define KIN 128     // input feature dim for P/A
#define ALPHA 0.2f
#define LDSPAD 8    // row pad (halves): stride 136 halves = 272 B, 16B-aligned

typedef __attribute__((ext_vector_type(16))) _Float16 v16h;
typedef __attribute__((ext_vector_type(8)))  _Float16 v8h;
typedef __attribute__((ext_vector_type(8)))  float    v8f;

// ---------------------------------------------------------------------------
// CAS-based float atomic max (values may be negative; init is -inf)
// ---------------------------------------------------------------------------
__device__ __forceinline__ void atomicMaxFloat(float* addr, float val) {
    unsigned int* ua = (unsigned int*)addr;
    unsigned int old = *ua;
    while (__uint_as_float(old) < val) {
        unsigned int assumed = old;
        old = atomicCAS(ua, assumed, __float_as_uint(val));
        if (old == assumed) break;
    }
}

// ---------------------------------------------------------------------------
// Fused 4-way projection GEMM:  Y_j = X @ W_j + b_j  (j = blockIdx.y in 0..3)
// X: [M,128] f32, W: [128,128] f32 row-major, Y: [M,128] f32.
// Block = 256 threads = 8 waves; block computes a 16(M) x 128(N) tile; wave w
// owns a 16x16 N-subtile; K in steps of 32 via v_wmma_f32_16x16x32_f16.
// W is staged TRANSPOSED in LDS so both A and B fragments are contiguous
// 16-byte segments -> ds_load_b128 (no scalar u16 gathers).
// ---------------------------------------------------------------------------
__global__ __launch_bounds__(256)
void gemm_proj4(const float* __restrict__ X,
                const float* W0, const float* W1, const float* W2, const float* W3,
                const float* b0, const float* b1, const float* b2, const float* b3,
                float* Y0, float* Y1, float* Y2, float* Y3, int M)
{
    const float* W; const float* bias; float* Y;
    switch (blockIdx.y) {
        case 0:  W = W0; bias = b0; Y = Y0; break;
        case 1:  W = W1; bias = b1; Y = Y1; break;
        case 2:  W = W2; bias = b2; Y = Y2; break;
        default: W = W3; bias = b3; Y = Y3; break;
    }

    __shared__ _Float16 Xs [16][KIN + LDSPAD];   // A tile, row-major (M x K)
    __shared__ _Float16 WsT[HD][KIN + LDSPAD];   // B, transposed (N x K)

    const int tid = threadIdx.x;                 // 0..255
    const int m0  = blockIdx.x * 16;

    // stage X tile (16x128) as f16
    for (int i = tid; i < 16 * KIN; i += 256) {
        int r = i >> 7, c = i & 127;
        float v = (m0 + r < M) ? X[(size_t)(m0 + r) * KIN + c] : 0.0f;
        Xs[r][c] = (_Float16)v;
    }
    // stage W (128x128) transposed as f16: WsT[n][k] = W[k*128+n]
    for (int i = tid; i < KIN * HD; i += 256) {
        int k = i >> 7, n = i & 127;
        WsT[n][k] = (_Float16)W[i];
    }
    __syncthreads();

    const int wave = tid >> 5;                   // 0..7 -> N tile index
    const int lane = tid & 31;
    const int n0   = wave * 16;
    const int grp  = lane >> 4;                  // half-wave group
    const int lm   = lane & 15;                  // A: M row ; B/C: N column

    const _Float16* xrow = &Xs[lm][0];
    const _Float16* wrow = &WsT[n0 + lm][0];

    v8f acc = {};
    #pragma unroll
    for (int k0 = 0; k0 < KIN; k0 += 32) {
        // A 16-bit 16x32 layout: grp0 halves -> K {0..7,16..23}, grp1 -> {8..15,24..31}
        v8h a0 = *(const v8h*)(xrow + k0 + 8 * grp);        // K  k0+8g   .. +7
        v8h a1 = *(const v8h*)(xrow + k0 + 16 + 8 * grp);   // K  k0+16+8g.. +7
        // B 16-bit 32x16 layout: grp0 halves -> K {0..7,16..23}, grp1 -> {8..15,24..31}
        v8h bb0 = *(const v8h*)(wrow + k0 + 8 * grp);
        v8h bb1 = *(const v8h*)(wrow + k0 + 16 + 8 * grp);
        v16h afrag = __builtin_shufflevector(a0, a1, 0,1,2,3,4,5,6,7,8,9,10,11,12,13,14,15);
        v16h bfrag = __builtin_shufflevector(bb0, bb1, 0,1,2,3,4,5,6,7,8,9,10,11,12,13,14,15);
        acc = __builtin_amdgcn_wmma_f32_16x16x32_f16(
            /*neg_a=*/false, afrag, /*neg_b=*/false, bfrag,
            /*c_mod=*/(short)0, acc, /*reuse_a=*/false, /*reuse_b=*/false);
    }

    // C/D layout: VGPR i, lanes 0-15 -> M=i, lanes 16-31 -> M=i+8; N = lane%16
    #pragma unroll
    for (int i = 0; i < 8; ++i) {
        int m = i + 8 * grp;
        int n = n0 + lm;
        if (m0 + m < M)
            Y[(size_t)(m0 + m) * HD + n] = acc[i] + bias[n];
    }
}

// tiny state projection: [1,64] @ [64,128] + b
__global__ void gemm_state(const float* __restrict__ x, const float* __restrict__ W,
                           const float* __restrict__ b, float* __restrict__ y)
{
    int n = threadIdx.x;                     // 128 threads
    float s = b[n];
    #pragma unroll 8
    for (int k = 0; k < 64; ++k) s += x[k] * W[k * HD + n];
    y[n] = s;
}

// ---------------------------------------------------------------------------
// per-node attention logits: out[n*H+h] = dot(Wh[n,h,:], a[h,:])
// ---------------------------------------------------------------------------
__global__ void rowdot(const float* __restrict__ Wh, const float* __restrict__ a,
                       float* __restrict__ out, int NH)
{
    int i = blockIdx.x * blockDim.x + threadIdx.x;
    if (i >= NH) return;
    int h = i & (H - 1);
    const float* row = Wh + (size_t)i * DD;  // (n*H+h)*D
    const float* av  = a + h * DD;
    float s = 0.0f;
    #pragma unroll 8
    for (int d = 0; d < DD; ++d) s += row[d] * av[d];
    out[i] = s;
}

__global__ void fillf(float* __restrict__ p, float v, size_t n)
{
    size_t i = (size_t)blockIdx.x * blockDim.x + threadIdx.x;
    size_t st = (size_t)gridDim.x * blockDim.x;
    for (; i < n; i += st) p[i] = v;
}

// pass 1: e = leakyrelu(es[src]+ed[dst]); store; segment-max into m
__global__ void edge_logits_max(const float* __restrict__ es, const float* __restrict__ ed,
                                const int* __restrict__ src, const int* __restrict__ dst,
                                float* __restrict__ ebuf, float* __restrict__ m, int E)
{
    int i = blockIdx.x * blockDim.x + threadIdx.x;   // over E*H
    if (i >= E * H) return;
    int e = i >> 2, h = i & 3;
    float v = es[src[e] * H + h] + ed[dst[e] * H + h];
    v = (v > 0.0f) ? v : ALPHA * v;
    ebuf[i] = v;
    atomicMaxFloat(&m[dst[e] * H + h], v);
}

// pass 2: w = exp(e - m[dst]); store; segment-sum into denom
__global__ void edge_exp_sum(const float* __restrict__ m, const int* __restrict__ dst,
                             float* __restrict__ ebuf, float* __restrict__ denom, int E)
{
    int i = blockIdx.x * blockDim.x + threadIdx.x;
    if (i >= E * H) return;
    int e = i >> 2, h = i & 3;
    float mm = m[dst[e] * H + h];
    if (mm < -1e37f) mm = 0.0f;                      // guard empty segments
    float w = __expf(ebuf[i] - mm);
    ebuf[i] = w;
    atomicAdd(&denom[dst[e] * H + h], w);
}

// pass 3: ft[dst] += Wh_src[src] * (w / denom[dst]);
// one thread per (edge, 4 features): float4 gather + 4 atomic f32 adds
__global__ void edge_aggregate(const float4* __restrict__ WhS4, const float* __restrict__ wbuf,
                               const float* __restrict__ denom,
                               const int* __restrict__ src, const int* __restrict__ dst,
                               float* __restrict__ ft, long long total /* E*32 */)
{
    long long i = (long long)blockIdx.x * blockDim.x + threadIdx.x;
    if (i >= total) return;
    int e = (int)(i >> 5);
    int q = (int)(i & 31);                   // quad index: features q*4 .. q*4+3
    int h = q >> 3;                          // (q*4)/32
    int s = src[e], t = dst[e];
    float a = wbuf[e * H + h] / denom[t * H + h];
    float4 v = WhS4[(size_t)s * 32 + q];
    float* p = ft + (size_t)t * HD + q * 4;
    atomicAdd(p + 0, v.x * a);
    atomicAdd(p + 1, v.y * a);
    atomicAdd(p + 2, v.z * a);
    atomicAdd(p + 3, v.w * a);
}

// out = relu(out + Wh_self)
__global__ void finalize_relu(const float* __restrict__ Wh, float* __restrict__ out, size_t n)
{
    size_t i = (size_t)blockIdx.x * blockDim.x + threadIdx.x;
    if (i >= n) return;
    float v = out[i] + Wh[i];
    out[i] = (v > 0.0f) ? v : 0.0f;
}

// ---------------------------------------------------------------------------
extern "C" void kernel_launch(void* const* d_in, const int* in_sizes, int n_in,
                              void* d_out, int out_size, void* d_ws, size_t ws_size,
                              hipStream_t stream)
{
    const float* feat_P     = (const float*)d_in[0];
    const float* feat_A     = (const float*)d_in[1];
    const float* feat_state = (const float*)d_in[2];
    const float* W_P   = (const float*)d_in[3];  const float* b_P   = (const float*)d_in[4];
    const float* W_A   = (const float*)d_in[5];  const float* b_A   = (const float*)d_in[6];
    const float* W_p2p = (const float*)d_in[7];  const float* b_p2p = (const float*)d_in[8];
    const float* W_p2a = (const float*)d_in[9];  const float* b_p2a = (const float*)d_in[10];
    const float* W_a2p = (const float*)d_in[11]; const float* b_a2p = (const float*)d_in[12];
    const float* W_a2a = (const float*)d_in[13]; const float* b_a2a = (const float*)d_in[14];
    const float* W_p2s = (const float*)d_in[15]; const float* b_p2s = (const float*)d_in[16];
    const float* W_a2s = (const float*)d_in[17]; const float* b_a2s = (const float*)d_in[18];
    const float* W_in  = (const float*)d_in[19]; const float* b_in  = (const float*)d_in[20];
    const float* at_p2p_s = (const float*)d_in[21]; const float* at_p2p_d = (const float*)d_in[22];
    const float* at_p2a_s = (const float*)d_in[23]; const float* at_p2a_d = (const float*)d_in[24];
    const float* at_a2p_s = (const float*)d_in[25]; const float* at_a2p_d = (const float*)d_in[26];
    const float* at_a2a_s = (const float*)d_in[27]; const float* at_a2a_d = (const float*)d_in[28];
    const float* at_p2s_s = (const float*)d_in[29]; const float* at_p2s_d = (const float*)d_in[30];
    const float* at_a2s_s = (const float*)d_in[31]; const float* at_a2s_d = (const float*)d_in[32];
    const int* src_p2p = (const int*)d_in[33]; const int* dst_p2p = (const int*)d_in[34];
    const int* src_p2a = (const int*)d_in[35]; const int* dst_p2a = (const int*)d_in[36];
    const int* src_a2p = (const int*)d_in[37]; const int* dst_a2p = (const int*)d_in[38];
    const int* src_a2a = (const int*)d_in[39]; const int* dst_a2a = (const int*)d_in[40];
    const int* src_p2s = (const int*)d_in[41]; const int* dst_p2s = (const int*)d_in[42];
    const int* src_a2s = (const int*)d_in[43]; const int* dst_a2s = (const int*)d_in[44];

    const int NP = in_sizes[0] / KIN;
    const int NA = in_sizes[1] / KIN;

    // ---- workspace carve-up (floats) ----
    float* ws = (float*)d_ws;
    size_t o = 0;
    float* Wh_P   = ws + o; o += (size_t)NP * HD;
    float* Wh_p2p = ws + o; o += (size_t)NP * HD;
    float* Wh_p2a = ws + o; o += (size_t)NP * HD;
    float* Wh_p2s = ws + o; o += (size_t)NP * HD;
    float* Wh_A   = ws + o; o += (size_t)NA * HD;
    float* Wh_a2p = ws + o; o += (size_t)NA * HD;
    float* Wh_a2a = ws + o; o += (size_t)NA * HD;
    float* Wh_a2s = ws + o; o += (size_t)NA * HD;
    float* Wh_in  = ws + o; o += HD;
    const int Nmax = (NP > NA) ? NP : NA;
    float* es_buf = ws + o; o += (size_t)Nmax * H;
    float* ed_buf = ws + o; o += (size_t)Nmax * H;
    float* m_buf  = ws + o; o += (size_t)Nmax * H;
    float* d_buf  = ws + o; o += (size_t)Nmax * H;
    int Emax = 0;
    { const int eix[6] = {33, 35, 37, 39, 41, 43};
      for (int t = 0; t < 6; ++t) if (in_sizes[eix[t]] > Emax) Emax = in_sizes[eix[t]]; }
    float* e_buf  = ws + o; o += (size_t)Emax * H;

    // ---- 1) projections (WMMA GEMMs) ----
    dim3 gP((NP + 15) / 16, 4), gA((NA + 15) / 16, 4);
    gemm_proj4<<<gP, 256, 0, stream>>>(feat_P, W_P, W_p2p, W_p2a, W_p2s,
                                       b_P, b_p2p, b_p2a, b_p2s,
                                       Wh_P, Wh_p2p, Wh_p2a, Wh_p2s, NP);
    gemm_proj4<<<gA, 256, 0, stream>>>(feat_A, W_A, W_a2p, W_a2a, W_a2s,
                                       b_A, b_a2p, b_a2a, b_a2s,
                                       Wh_A, Wh_a2p, Wh_a2a, Wh_a2s, NA);
    gemm_state<<<1, 128, 0, stream>>>(feat_state, W_in, b_in, Wh_in);

    // ---- 2) zero ft accumulators (live in d_out) ----
    size_t outN = (size_t)out_size;
    fillf<<<(unsigned)((outN + 255) / 256), 256, 0, stream>>>((float*)d_out, 0.0f, outN);

    float* ft_P = (float*)d_out;
    float* ft_A = ft_P + (size_t)NP * HD;
    float* ft_S = ft_A + (size_t)NA * HD;

    // ---- 3) six edge types, sequentially ----
    struct Et { const float* WhS; const float* WhD; const float* as_; const float* ad_;
                const int* src; const int* dst; int E; int Ns; int Nd; float* ft; };
    const Et ets[6] = {
        { Wh_p2p, Wh_P,  at_p2p_s, at_p2p_d, src_p2p, dst_p2p, in_sizes[33], NP, NP, ft_P },
        { Wh_a2p, Wh_P,  at_a2p_s, at_a2p_d, src_a2p, dst_a2p, in_sizes[37], NA, NP, ft_P },
        { Wh_p2a, Wh_A,  at_p2a_s, at_p2a_d, src_p2a, dst_p2a, in_sizes[35], NP, NA, ft_A },
        { Wh_a2a, Wh_A,  at_a2a_s, at_a2a_d, src_a2a, dst_a2a, in_sizes[39], NA, NA, ft_A },
        { Wh_p2s, Wh_in, at_p2s_s, at_p2s_d, src_p2s, dst_p2s, in_sizes[41], NP, 1,  ft_S },
        { Wh_a2s, Wh_in, at_a2s_s, at_a2s_d, src_a2s, dst_a2s, in_sizes[43], NA, 1,  ft_S },
    };
    for (int t = 0; t < 6; ++t) {
        const Et& e = ets[t];
        const int nsH = e.Ns * H, ndH = e.Nd * H, eH = e.E * H;
        rowdot<<<(nsH + 255) / 256, 256, 0, stream>>>(e.WhS, e.as_, es_buf, nsH);
        rowdot<<<(ndH + 255) / 256, 256, 0, stream>>>(e.WhD, e.ad_, ed_buf, ndH);
        fillf <<<(ndH + 255) / 256, 256, 0, stream>>>(m_buf, -INFINITY, (size_t)ndH);
        fillf <<<(ndH + 255) / 256, 256, 0, stream>>>(d_buf, 0.0f,      (size_t)ndH);
        edge_logits_max<<<(eH + 255) / 256, 256, 0, stream>>>(es_buf, ed_buf, e.src, e.dst,
                                                              e_buf, m_buf, e.E);
        edge_exp_sum   <<<(eH + 255) / 256, 256, 0, stream>>>(m_buf, e.dst, e_buf, d_buf, e.E);
        const long long tot = (long long)e.E * 32;   // (edge, float4-quad) pairs
        edge_aggregate<<<(unsigned)((tot + 255) / 256), 256, 0, stream>>>(
            (const float4*)e.WhS, e_buf, d_buf, e.src, e.dst, e.ft, tot);
    }

    // ---- 4) out = relu(ft + Wh_self) ----
    size_t nPf = (size_t)NP * HD, nAf = (size_t)NA * HD;
    finalize_relu<<<(unsigned)((nPf + 255) / 256), 256, 0, stream>>>(Wh_P,  ft_P, nPf);
    finalize_relu<<<(unsigned)((nAf + 255) / 256), 256, 0, stream>>>(Wh_A,  ft_A, nAf);
    finalize_relu<<<1, 128, 0, stream>>>(Wh_in, ft_S, (size_t)HD);
}
